// D_DiffDR_88837103551019
// MI455X (gfx1250) — matrix operand order repaired and checked by reading, not compiled, not run on hardware
//
#include <hip/hip_runtime.h>
#include <hip/hip_bf16.h>

typedef __attribute__((ext_vector_type(16))) _Float16 v16h;
typedef __attribute__((ext_vector_type(8)))  _Float16 v8h;
typedef __attribute__((ext_vector_type(8)))  float    v8f;

#define T_STEPS 64
#define GRES    97
#define G2      9409          // 97*97
#define NTOT    912673        // 97^3
#define KPAD    72            // padded K stride in halves (bank-conflict-free)
#define TILE_N  256
#define TWO_PI  6.283185307179586f

// ---- fragment loads from LDS ---------------------------------------------
// A (16-bit, 16x32): lane-half lh: halves 0..7 -> K = kb..kb+7, halves 8..15 -> K = kb+16..kb+23
__device__ __forceinline__ v16h ldA(const _Float16* row, int kb) {
  v8h lo = *(const v8h*)(row + kb);
  v8h hi = *(const v8h*)(row + kb + 16);
  v16h r;
#pragma unroll
  for (int i = 0; i < 8; ++i) { r[i] = lo[i]; r[i + 8] = hi[i]; }
  return r;
}
// B (16-bit, 32x16): lane-half lh: halves 0..7 -> K = kb..kb+7, halves 8..15 -> K = kb+16..kb+23
__device__ __forceinline__ v16h ldB(const _Float16* row, int kb) {
  v8h lo = *(const v8h*)(row + kb);
  v8h hi = *(const v8h*)(row + kb + 16);
  v16h r;
#pragma unroll
  for (int i = 0; i < 8; ++i) { r[i] = lo[i]; r[i + 8] = hi[i]; }
  return r;
}

// ---- block reduction (blockDim.x == 256, wave32) --------------------------
__device__ __forceinline__ float block_reduce(float v, float* sm) {
#pragma unroll
  for (int o = 16; o > 0; o >>= 1) v += __shfl_down(v, o, 32);
  const int lane = threadIdx.x & 31;
  const int w = threadIdx.x >> 5;
  if (lane == 0) sm[w] = v;
  __syncthreads();
  float r = 0.0f;
  if (threadIdx.x < 8) r = sm[threadIdx.x];
  if (w == 0) {
#pragma unroll
    for (int o = 4; o > 0; o >>= 1) r += __shfl_down(r, o, 32);
  }
  __syncthreads();
  return r;
}

// ===========================================================================
// Kernel 1: sdf/flow via WMMA f16 GEMM + entropy / |flow| partial sums
// ===========================================================================
__global__ void __launch_bounds__(256)
k_eval(const float* __restrict__ times, const float* __restrict__ ssdf,
       const float* __restrict__ poly,  const float* __restrict__ fl,
       const float* __restrict__ fm,    const float* __restrict__ fh,
       float* __restrict__ out, float* __restrict__ acc)
{
  __shared__ alignas(16) _Float16 bS[T_STEPS * KPAD];   // sdf basis  [t][k]
  __shared__ alignas(16) _Float16 bF[T_STEPS * KPAD];   // flow basis [t][k]
  __shared__ alignas(16) _Float16 cH[TILE_N * KPAD];    // coefs      [v][k]
  __shared__ float red[8];

  const int tid  = threadIdx.x;
  const int base = blockIdx.x * TILE_N;

  for (int i = tid; i < TILE_N * KPAD; i += 256) cH[i] = (_Float16)0.0f;
  for (int i = tid; i < T_STEPS * KPAD; i += 256) { bS[i] = (_Float16)0.0f; bF[i] = (_Float16)0.0f; }
  __syncthreads();

  // --- basis matrices (shared by every vertex tile) ---
  {
    const int t  = tid & 63;
    const int kc = tid >> 6;                 // 4 chunks of 16 k's
    const float tv = times[t];
#pragma unroll
    for (int i = 0; i < 16; ++i) {
      const int k = kc * 16 + i;
      float S = 0.0f, F = 0.0f;
      if (k == 0)      { S = 1.0f;           F = 0.0f; }
      else if (k == 1) { S = tv;             F = 1.0f; }
      else if (k == 2) { S = tv * tv;        F = 2.0f * tv; }
      else if (k == 3) { S = tv * tv * tv;   F = 3.0f * tv * tv; }
      else if (k < 22) {                      // cos terms, freq = k-3
        const float w = TWO_PI * (float)(k - 3);
        float sn, cs; __sincosf(w * tv, &sn, &cs);
        S = cs; F = -w * sn;
      } else if (k < 40) {                    // sin terms, freq = k-21
        const float w = TWO_PI * (float)(k - 21);
        float sn, cs; __sincosf(w * tv, &sn, &cs);
        S = sn; F = w * cs;
      }
      bS[t * KPAD + k] = (_Float16)S;
      bF[t * KPAD + k] = (_Float16)F;
    }
  }

  // --- stage coefficients (transposed, f16) ---
  { const int n = base + tid;
    if (n < NTOT) cH[tid * KPAD + 0] = (_Float16)ssdf[n]; }
  for (int idx = tid; idx < TILE_N * 3; idx += 256) {
    const int v = idx / 3, d = idx - v * 3, n = base + v;
    if (n < NTOT) cH[v * KPAD + 1 + d] = (_Float16)poly[(size_t)n * 3 + d];
  }
  for (int idx = tid; idx < TILE_N * 6; idx += 256) {
    const int v = idx / 6, j = idx - v * 6, n = base + v;
    if (n < NTOT) cH[v * KPAD + ((j & 1) ? 22 : 4) + (j >> 1)] = (_Float16)fl[(size_t)n * 6 + j];
  }
  for (int idx = tid; idx < TILE_N * 12; idx += 256) {
    const int v = idx / 12, j = idx - v * 12, n = base + v;
    if (n < NTOT) cH[v * KPAD + ((j & 1) ? 25 : 7) + (j >> 1)] = (_Float16)fm[(size_t)n * 12 + j];
  }
  for (int idx = tid; idx < TILE_N * 18; idx += 256) {
    const int v = idx / 18, j = idx - v * 18, n = base + v;
    if (n < NTOT) cH[v * KPAD + ((j & 1) ? 31 : 13) + (j >> 1)] = (_Float16)fh[(size_t)n * 18 + j];
  }
  __syncthreads();

  const int lane = tid & 31;
  const int wv   = tid >> 5;
  const int m    = lane & 15;
  const int lh   = lane >> 4;

  float entSum = 0.0f, ttvSum = 0.0f;

#pragma unroll
  for (int sub = 0; sub < 2; ++sub) {
    const int ln = wv * 32 + sub * 16;        // 16-vertex column tile
    v8f accS[4] = {};
    v8f accF[4] = {};

#pragma unroll
    for (int ks = 0; ks < 2; ++ks) {          // K = 0..31, 32..63 (40..63 zero)
      const v16h bfrag = ldB(&cH[(ln + m) * KPAD], ks * 32 + lh * 8);
      const int kbA = ks * 32 + lh * 8;
#pragma unroll
      for (int mt = 0; mt < 4; ++mt) {        // time tiles 0..3
        const int t = mt * 16 + m;
        const v16h aS = ldA(&bS[t * KPAD], kbA);
        const v16h aF = ldA(&bF[t * KPAD], kbA);
        accS[mt] = __builtin_amdgcn_wmma_f32_16x16x32_f16(false, aS, false, bfrag,
                                                          (short)0, accS[mt], false, false);
        accF[mt] = __builtin_amdgcn_wmma_f32_16x16x32_f16(false, aF, false, bfrag,
                                                          (short)0, accF[mt], false, false);
      }
    }

    const int n = base + ln + m;              // D: N = lane%16
    if (n < NTOT) {
#pragma unroll
      for (int mt = 0; mt < 4; ++mt) {
#pragma unroll
        for (int v = 0; v < 8; ++v) {
          const int t = mt * 16 + lh * 8 + v; // D: M = v + 8*(lane/16)
          const float s = accS[mt][v];
          const float f = accF[mt][v];
          out[(size_t)t * NTOT + n] = s;
          out[(size_t)(T_STEPS + t) * NTOT + n] = f;
          // entropy of sigmoid(s): softplus(s) - s*sigmoid(s)
          const float ax = fabsf(s);
          const float e  = __expf(-ax);
          const float sp = __logf(1.0f + e) + fmaxf(s, 0.0f);
          const float p  = (s >= 0.0f) ? (1.0f / (1.0f + e)) : (e / (1.0f + e));
          entSum += sp - s * p;
          ttvSum += fabsf(f);
        }
      }
    }
  }

  const float eb = block_reduce(entSum, red);
  const float tb = block_reduce(ttvSum, red);
  if (tid == 0) {
    atomicAdd(&acc[0], eb * (1.0f / 58411072.0f));   // / (T*N)
    atomicAdd(&acc[2], tb * (1.0f / 58411072.0f));   // / (T*N)
  }
}

// ===========================================================================
// Kernel 2: eikonal + scene-flow smoothness from written sdf/flow
// ===========================================================================
__global__ void __launch_bounds__(256)
k_spatial(const float* __restrict__ out, float* __restrict__ acc)
{
  __shared__ float red[8];
  const size_t TN = (size_t)T_STEPS * NTOT;
  float eikSum = 0.0f, sfsSum = 0.0f;

  for (int n = blockIdx.x * 256 + threadIdx.x; n < NTOT; n += gridDim.x * 256) {
    const int x = n / G2;
    const int r = n - x * G2;
    const int y = r / GRES;
    const int z = r - y * GRES;
    const bool bx = x < GRES - 1, by = y < GRES - 1, bz = z < GRES - 1;
    const bool ball = bx && by && bz;
    for (int t = 0; t < T_STEPS; ++t) {
      const size_t idx = (size_t)t * NTOT + n;
      const float f = out[TN + idx];
      if (bx) { const float d = out[TN + idx + G2]   - f; sfsSum += d * d; }
      if (by) { const float d = out[TN + idx + GRES] - f; sfsSum += d * d; }
      if (bz) { const float d = out[TN + idx + 1]    - f; sfsSum += d * d; }
      if (ball) {
        const float s  = out[idx];
        const float gx = (out[idx + G2]   - s) * 96.0f;   // / HSP
        const float gy = (out[idx + GRES] - s) * 96.0f;
        const float gz = (out[idx + 1]    - s) * 96.0f;
        const float gn = sqrtf(gx * gx + gy * gy + gz * gz + 1e-12f);
        const float d  = gn - 1.0f;
        eikSum += d * d;
      }
    }
  }
  const float e = block_reduce(eikSum, red);
  const float s = block_reduce(sfsSum, red);
  if (threadIdx.x == 0) {
    atomicAdd(&acc[1], e * (1.0f / 56623104.0f));    // / (T*96^3)
    atomicAdd(&acc[4], s * (1.0f / 173426688.0f));   // / (3*T*96*97*97)
  }
}

// ===========================================================================
// Kernel 3: coefficient TV over the 5 input arrays (40 channels total)
// ===========================================================================
__device__ __forceinline__ float tv_arr(const float* __restrict__ p, int n, int C,
                                        bool bx, bool by, bool bz)
{
  float s = 0.0f;
  const float* p0 = p + (size_t)n * C;
  for (int j = 0; j < C; ++j) {
    const float c0 = p0[j];
    if (bx) s += fabsf(p0[(size_t)G2 * C + j]   - c0);
    if (by) s += fabsf(p0[(size_t)GRES * C + j] - c0);
    if (bz) s += fabsf(p0[C + j]                - c0);
  }
  return s;
}

__global__ void __launch_bounds__(256)
k_ctv(const float* __restrict__ ssdf, const float* __restrict__ poly,
      const float* __restrict__ fl,   const float* __restrict__ fm,
      const float* __restrict__ fh,   float* __restrict__ acc)
{
  __shared__ float red[8];
  float sum = 0.0f;
  for (int n = blockIdx.x * 256 + threadIdx.x; n < NTOT; n += gridDim.x * 256) {
    const int x = n / G2;
    const int r = n - x * G2;
    const int y = r / GRES;
    const int z = r - y * GRES;
    const bool bx = x < GRES - 1, by = y < GRES - 1, bz = z < GRES - 1;
    sum += tv_arr(ssdf, n, 1,  bx, by, bz);
    sum += tv_arr(poly, n, 3,  bx, by, bz);
    sum += tv_arr(fl,   n, 6,  bx, by, bz);
    sum += tv_arr(fm,   n, 12, bx, by, bz);
    sum += tv_arr(fh,   n, 18, bx, by, bz);
  }
  const float s = block_reduce(sum, red);
  if (threadIdx.x == 0)
    atomicAdd(&acc[3], s * (1.0f / 108391680.0f));   // / (3*96*97*97*40)
}

// ===========================================================================
__global__ void k_init(float* acc) { if (threadIdx.x < 8) acc[threadIdx.x] = 0.0f; }

__global__ void k_fin(const float* __restrict__ acc, float* __restrict__ out5)
{
  if (threadIdx.x < 5) out5[threadIdx.x] = acc[threadIdx.x]; // ent,eik,ttv,ctv,sfs
}

// ===========================================================================
extern "C" void kernel_launch(void* const* d_in, const int* in_sizes, int n_in,
                              void* d_out, int out_size, void* d_ws, size_t ws_size,
                              hipStream_t stream)
{
  const float* times = (const float*)d_in[0];
  const float* ssdf  = (const float*)d_in[1];
  const float* poly  = (const float*)d_in[2];
  const float* fl    = (const float*)d_in[3];
  const float* fm    = (const float*)d_in[4];
  const float* fh    = (const float*)d_in[5];
  float* out = (float*)d_out;
  float* acc = (float*)d_ws;

  const int blocks = (NTOT + TILE_N - 1) / TILE_N;   // 3566

  hipLaunchKernelGGL(k_init, dim3(1), dim3(32), 0, stream, acc);
  hipLaunchKernelGGL(k_eval, dim3(blocks), dim3(256), 0, stream,
                     times, ssdf, poly, fl, fm, fh, out, acc);
  hipLaunchKernelGGL(k_spatial, dim3(blocks), dim3(256), 0, stream, out, acc);
  hipLaunchKernelGGL(k_ctv, dim3(blocks), dim3(256), 0, stream,
                     ssdf, poly, fl, fm, fh, acc);
  hipLaunchKernelGGL(k_fin, dim3(1), dim3(32), 0, stream,
                     acc, out + (size_t)2 * T_STEPS * NTOT);
}
